// SelfAttention_80247168958892
// MI455X (gfx1250) — compile-verified, matrix operand order checked by reading, not run on hardware
//
#include <hip/hip_runtime.h>

// ---------------------------------------------------------------------------
// Trilinear self-attention (B=32, L=1024, H=1024, fp32 in/out) for gfx1250.
//
// Compute-bound: 2 x 68.7 GFLOP GEMMs -> v_wmma_f32_16x16x32_bf16.
// Two cheap streaming passes convert x to bf16 operand layouts (row-major
// xbf, qbf = bf16(x*w3), d-major xbfT); both GEMM kernels then load WMMA
// fragments DIRECTLY from global as 2x b128 per fragment. No LDS staging,
// no inner-loop barriers; per-batch operands (2 MB bf16) stay L2-resident
// across the 32 tile-workgroups that reuse them.
// Both GEMM kernels keep K as the single outer reduction loop with all
// per-wave output tiles live (acc[2][8] = 128 VGPRs) so no operand stream
// is loop-invariant -> no register-pressure hoisting/spills.
// Denominator = rowsum + exp(bias): no online-max needed.
// ---------------------------------------------------------------------------

#define BB   32
#define LL   1024
#define HH   1024
#define NEG_INF (-1.0e9f)

typedef __attribute__((ext_vector_type(4)))  __bf16 v4bf;
typedef __attribute__((ext_vector_type(8)))  __bf16 v8bf;
typedef __attribute__((ext_vector_type(16))) __bf16 v16bf;
typedef __attribute__((ext_vector_type(4)))  float  v4f;
typedef __attribute__((ext_vector_type(8)))  float  v8f;

// A/B bf16 fragment (16x16x32): per lane, elements 0..7 = K kb+0..7 and
// elements 8..15 = K kb+16..23 -> two contiguous 16-byte loads.
__device__ __forceinline__ v16bf load_frag16(const __bf16* __restrict__ p)
{
    v8bf lo = *(const v8bf*)(p);
    v8bf hi = *(const v8bf*)(p + 16);
    return __builtin_shufflevector(lo, hi, 0, 1, 2, 3, 4, 5, 6, 7,
                                           8, 9, 10, 11, 12, 13, 14, 15);
}

// ---------------------------------------------------------------------------
// Kernel 1: per-row a = x.w1, b = x.w2; rowsum := 0; xbf = bf16(x);
// qbf = bf16(x*w3). One wave per row, float4 streams.
// ---------------------------------------------------------------------------
__global__ __launch_bounds__(256)
void prep_kernel(const float* __restrict__ x,
                 const float* __restrict__ w1,
                 const float* __restrict__ w2,
                 const float* __restrict__ w3,
                 float*  __restrict__ aVec,
                 float*  __restrict__ bVec,
                 float*  __restrict__ rSum,
                 __bf16* __restrict__ xbf,
                 __bf16* __restrict__ qbf)
{
    const int wave = threadIdx.x >> 5;
    const int lane = threadIdx.x & 31;
    const int row  = blockIdx.x * 8 + wave;            // global (b*L + l)

    const v4f* xr = (const v4f*)(x + (size_t)row * HH);
    const v4f* W1 = (const v4f*)w1;
    const v4f* W2 = (const v4f*)w2;
    const v4f* W3 = (const v4f*)w3;
    v4bf* xo = (v4bf*)(xbf + (size_t)row * HH);
    v4bf* qo = (v4bf*)(qbf + (size_t)row * HH);

    float s1 = 0.f, s2 = 0.f;
    for (int c = lane; c < HH / 4; c += 32) {
        v4f xv = xr[c], a1 = W1[c], a2 = W2[c], a3 = W3[c];
        v4bf xb, qb;
        #pragma unroll
        for (int i = 0; i < 4; ++i) {
            s1 += xv[i] * a1[i];
            s2 += xv[i] * a2[i];
            xb[i] = (__bf16)xv[i];
            qb[i] = (__bf16)(xv[i] * a3[i]);
        }
        xo[c] = xb;
        qo[c] = qb;
    }
    #pragma unroll
    for (int m = 16; m >= 1; m >>= 1) {
        s1 += __shfl_xor(s1, m, 32);
        s2 += __shfl_xor(s2, m, 32);
    }
    if (lane == 0) {
        aVec[row] = s1;
        bVec[row] = s2;
        rSum[row] = 0.f;                               // for score-kernel atomics
    }
}

// ---------------------------------------------------------------------------
// Kernel 2: xbfT[b][h][l] = bf16(x[b][l][h]) via 32x32 LDS tile.
// ---------------------------------------------------------------------------
__global__ __launch_bounds__(256)
void transpose_kernel(const __bf16* __restrict__ xbf,
                      __bf16* __restrict__ xbfT)
{
    const int b  = blockIdx.x >> 10;
    const int l0 = ((blockIdx.x >> 5) & 31) * 32;
    const int h0 = (blockIdx.x & 31) * 32;

    __shared__ __bf16 T[32][40];                        // pad keeps b64 aligned

    const int tid = threadIdx.x;
    {   // load 32(l) x 32(h) tile, coalesced b64 per thread
        int r = tid >> 3, c4 = (tid & 7) * 4;
        v4bf v = *(const v4bf*)&xbf[((size_t)(b * LL + l0 + r)) * HH + h0 + c4];
        *(v4bf*)&T[r][c4] = v;
    }
    __syncthreads();
    {   // store transposed, coalesced b64 per thread
        int hr = tid >> 3, l4 = (tid & 7) * 4;
        v4bf w;
        #pragma unroll
        for (int i = 0; i < 4; ++i) w[i] = T[l4 + i][hr];
        *(v4bf*)&xbfT[((size_t)(b * HH + h0 + hr)) * LL + l0 + l4] = w;
    }
}

// ---------------------------------------------------------------------------
// Kernel 3: E[b,i,j] = exp(masked(a_i + b_j + q_i . x_j)); rowsum atomics.
// WG = (batch, 32-row query tile), 8 waves. Wave w owns contiguous j-tiles
// [w*8, w*8+8) and keeps all 16 output tiles live (acc[2][8]).
// K is the single outer loop: per k0 -> 2 A-frag + 8 B-frag b128 pairs,
// 16 WMMAs. Constant jj*16*H byte offsets fold into the load ioffset.
// ---------------------------------------------------------------------------
__global__ __launch_bounds__(256)
void score_kernel(const __bf16* __restrict__ qbf,
                  const __bf16* __restrict__ xbf,
                  const float*  __restrict__ qm,
                  const float*  __restrict__ aVec,
                  const float*  __restrict__ bVec,
                  float*  __restrict__ rSum,
                  __bf16* __restrict__ E)
{
    const int b  = blockIdx.x >> 5;                    // 32 i-tiles per batch
    const int i0 = (blockIdx.x & 31) * 32;

    __shared__ float aS[32], qmS[32];
    const int tid = threadIdx.x;
    if (tid < 32) {
        aS[tid]  = aVec[b * LL + i0 + tid];
        qmS[tid] = qm  [b * LL + i0 + tid];
    }
    __syncthreads();

    const int wave = tid >> 5;
    const int lane = tid & 31;
    const int half = lane >> 4;
    const int ln   = lane & 15;
    const int kb   = half * 8;

    const __bf16* qrow0 = qbf + ((size_t)(b * LL + i0 + ln)) * HH + kb;
    const __bf16* qrow1 = qrow0 + (size_t)16 * HH;
    // wave w's 8 key tiles start at row wave*128; tile jj at +jj*16 rows.
    const __bf16* krow  = xbf + ((size_t)(b * LL + wave * 128 + ln)) * HH + kb;

    v8f acc[2][8];
    #pragma unroll
    for (int tI = 0; tI < 2; ++tI)
        #pragma unroll
        for (int jj = 0; jj < 8; ++jj) acc[tI][jj] = {};

    for (int k0 = 0; k0 < HH; k0 += 32) {
        v16bf a0 = load_frag16(qrow0 + k0);
        v16bf a1 = load_frag16(qrow1 + k0);
        #pragma unroll
        for (int jj = 0; jj < 8; ++jj) {
            v16bf bf = load_frag16(krow + (size_t)(jj * 16) * HH + k0);
            acc[0][jj] = __builtin_amdgcn_wmma_f32_16x16x32_bf16(
                             false, a0, false, bf, (short)0, acc[0][jj], false, false);
            acc[1][jj] = __builtin_amdgcn_wmma_f32_16x16x32_bf16(
                             false, a1, false, bf, (short)0, acc[1][jj], false, false);
        }
    }

    // Epilogue: masks, exp, bf16 E store, local row-sum accumulation.
    float rsum[2][8];
    #pragma unroll
    for (int tI = 0; tI < 2; ++tI)
        #pragma unroll
        for (int r = 0; r < 8; ++r) rsum[tI][r] = 0.f;

    #pragma unroll
    for (int jj = 0; jj < 8; ++jj) {
        const int   jcol = (wave * 8 + jj) * 16 + ln;
        const float bj   = bVec[b * LL + jcol];
        const float qmj  = qm  [b * LL + jcol];
        #pragma unroll
        for (int tI = 0; tI < 2; ++tI) {
            #pragma unroll
            for (int r = 0; r < 8; ++r) {
                int   m    = tI * 16 + half * 8 + r;    // C/D: M = r + 8*half
                int   irow = i0 + m;
                float s    = acc[tI][jj][r] + aS[m] + bj;
                if (irow == jcol || qmS[m] == 0.f || qmj == 0.f) s = NEG_INF;
                float e = __expf(s);                    // exp(-1e9) -> 0
                E[((size_t)b * LL + irow) * LL + jcol] = (__bf16)e;
                rsum[tI][r] += e;
            }
        }
    }

    // Row sums: reduce across the 16 lanes of each half, then one atomic each.
    #pragma unroll
    for (int tI = 0; tI < 2; ++tI) {
        #pragma unroll
        for (int r = 0; r < 8; ++r) {
            float v = rsum[tI][r];
            #pragma unroll
            for (int m = 8; m >= 1; m >>= 1) v += __shfl_xor(v, m, 32);
            if (ln == 0)
                atomicAdd(&rSum[b * LL + i0 + tI * 16 + half * 8 + r], v);
        }
    }
}

// ---------------------------------------------------------------------------
// Kernel 4: agg[b,i,d] = (sum_j E[b,i,j] * x[b,j,d]) * qm_i/(rowsum_i+e^bias)
// WG = (batch, 32-row tile); wave w owns d-slice [w*128, w*128+128).
// Per 32-K slab: 4 A-frag + 16 B-frag b128 loads, 16 WMMAs. No LDS staging.
// ---------------------------------------------------------------------------
__global__ __launch_bounds__(256)
void agg_kernel(const __bf16* __restrict__ xbfT,
                const __bf16* __restrict__ E,
                const float*  __restrict__ qm,
                const float*  __restrict__ bias,
                const float*  __restrict__ rSum,
                float* __restrict__ out)
{
    const int b  = blockIdx.x >> 5;
    const int i0 = (blockIdx.x & 31) * 32;

    __shared__ float scaleS[32];
    const int tid = threadIdx.x;
    if (tid < 32) {
        float denom = rSum[b * LL + i0 + tid] + __expf(bias[0]);
        scaleS[tid] = qm[b * LL + i0 + tid] / denom;
    }
    __syncthreads();

    const int wave = tid >> 5;
    const int lane = tid & 31;
    const int half = lane >> 4;
    const int ln   = lane & 15;
    const int kb   = half * 8;

    const __bf16* erow0 = E + ((size_t)(b * LL + i0 + ln)) * LL + kb;
    const __bf16* erow1 = erow0 + (size_t)16 * LL;
    const __bf16* vbase = xbfT + ((size_t)(b * HH + wave * 128 + ln)) * LL + kb;

    v8f acc[2][8];
    #pragma unroll
    for (int tI = 0; tI < 2; ++tI)
        #pragma unroll
        for (int dt = 0; dt < 8; ++dt) acc[tI][dt] = {};

    for (int j0 = 0; j0 < LL; j0 += 32) {
        v16bf a0 = load_frag16(erow0 + j0);
        v16bf a1 = load_frag16(erow1 + j0);
        #pragma unroll
        for (int dt = 0; dt < 8; ++dt) {
            // B[k=j][n=d] row d is contiguous in j in xbfT.
            v16bf bf = load_frag16(vbase + (size_t)(dt * 16) * LL + j0);
            acc[0][dt] = __builtin_amdgcn_wmma_f32_16x16x32_bf16(
                             false, a0, false, bf, (short)0, acc[0][dt], false, false);
            acc[1][dt] = __builtin_amdgcn_wmma_f32_16x16x32_bf16(
                             false, a1, false, bf, (short)0, acc[1][dt], false, false);
        }
    }

    // Scale + store fp32 (lanes within a half write 64B-contiguous runs).
    #pragma unroll
    for (int tI = 0; tI < 2; ++tI) {
        #pragma unroll
        for (int dt = 0; dt < 8; ++dt) {
            const int d = wave * 128 + dt * 16 + ln;
            #pragma unroll
            for (int r = 0; r < 8; ++r) {
                int m = tI * 16 + half * 8 + r;
                out[((size_t)b * LL + i0 + m) * HH + d] = acc[tI][dt][r] * scaleS[m];
            }
        }
    }
}

// ---------------------------------------------------------------------------
// Launch: prep -> transpose -> score -> agg (stream-ordered).
// Workspace: a|b|rowsum (3 * B*L f32) , xbf , qbf , xbfT (B*L*H bf16 each),
// E (B*L*L bf16)  ->  ~256.4 MB.
// ---------------------------------------------------------------------------
extern "C" void kernel_launch(void* const* d_in, const int* in_sizes, int n_in,
                              void* d_out, int out_size, void* d_ws, size_t ws_size,
                              hipStream_t stream)
{
    const float* x    = (const float*)d_in[0];   // [B, L, H]
    const float* qmsk = (const float*)d_in[1];   // [B, L]
    const float* w1   = (const float*)d_in[2];   // [H]
    const float* w2   = (const float*)d_in[3];   // [H]
    const float* w3   = (const float*)d_in[4];   // [H]
    const float* bias = (const float*)d_in[5];   // [1]
    float* out = (float*)d_out;                  // [B, L, H]

    float*  aVec = (float*)d_ws;
    float*  bVec = aVec + (size_t)BB * LL;
    float*  rSum = bVec + (size_t)BB * LL;
    __bf16* xbf  = (__bf16*)(rSum + (size_t)BB * LL);
    __bf16* qbf  = xbf  + (size_t)BB * LL * HH;
    __bf16* xbfT = qbf  + (size_t)BB * LL * HH;
    __bf16* E    = xbfT + (size_t)BB * LL * HH;

    (void)in_sizes; (void)n_in; (void)out_size; (void)ws_size;

    prep_kernel<<<(BB * LL) / 8, 256, 0, stream>>>(x, w1, w2, w3,
                                                   aVec, bVec, rSum, xbf, qbf);
    transpose_kernel<<<BB * (LL / 32) * (HH / 32), 256, 0, stream>>>(xbf, xbfT);
    score_kernel<<<BB * (LL / 32), 256, 0, stream>>>(qbf, xbf, qmsk,
                                                     aVec, bVec, rSum, E);
    agg_kernel<<<BB * (LL / 32), 256, 0, stream>>>(xbfT, E, qmsk, bias, rSum, out);
}
